// OTTLinear_45724221833506
// MI455X (gfx1250) — matrix operand order repaired and checked by reading, compiled, not run on hardware
//
#include <hip/hip_runtime.h>
#include <hip/hip_bf16.h>

typedef __attribute__((ext_vector_type(8)))  __bf16 v8bf;
typedef __attribute__((ext_vector_type(16))) __bf16 v16bf;
typedef __attribute__((ext_vector_type(8)))  float  v8f;

#define KDIM 4096
#define BATCH 8192

// CDNA5 async global->LDS copy path (ASYNCcnt tracked), guarded so the host
// pass and older toolchains fall back to synchronous LDS staging.
#if defined(__AMDGCN__) && __has_builtin(__builtin_amdgcn_global_load_async_to_lds_b128)
#define ASYNC_CP 1
#else
#define ASYNC_CP 0
#endif

#if ASYNC_CP
// Parameter types per clang diagnostics: (v4i AS1*, v4i AS3*, imm, imm)
typedef int v4i_vs __attribute__((vector_size(16)));
#define AS1_V4I(p) ((__attribute__((address_space(1))) v4i_vs*)(p))
#define AS3_V4I(p) ((__attribute__((address_space(3))) v4i_vs*)(p))
#if __has_builtin(__builtin_amdgcn_s_wait_asynccnt)
#define WAIT_ASYNC(n) __builtin_amdgcn_s_wait_asynccnt((short)(n))
#else
#define WAIT_ASYNC(n) asm volatile("s_wait_asynccnt %0" ::"i"(n) : "memory")
#endif
#endif

// ---------------------------------------------------------------------------
// Phase 1: batched Cayley transform. One block per 16x16 matrix, thread (i,j).
// Computes Q = (I+A)^-1 (I-A) == (I-A)(I+A)^-1 (they commute) via pivot-free
// Gauss-Jordan on augmented [M|N], valid since I+A has SPD symmetric part.
// ---------------------------------------------------------------------------
__global__ void cayley_kernel(const float* __restrict__ P, float* __restrict__ Q) {
  __shared__ float Ms[16][17];
  __shared__ float Ns[16][17];
  __shared__ float fs[16];
  const int n = blockIdx.x;
  const int i = threadIdx.x >> 4, j = threadIdx.x & 15;
  const float p_ij = P[n * 256 + i * 16 + j];
  const float p_ji = P[n * 256 + j * 16 + i];
  const float a = (j > i ? p_ij : 0.f) - (i > j ? p_ji : 0.f);  // A = triu(P,1) - triu(P,1)^T
  Ms[i][j] = (i == j ? 1.f : 0.f) + a;   // M = I + A
  Ns[i][j] = (i == j ? 1.f : 0.f) - a;   // N = I - A
  __syncthreads();
  for (int k = 0; k < 16; ++k) {
    const float piv = Ms[k][k];
    __syncthreads();
    if (i == k) { Ms[k][j] /= piv; Ns[k][j] /= piv; }
    __syncthreads();
    if (j == k) fs[i] = Ms[i][k];
    __syncthreads();
    if (i != k) {
      const float f = fs[i];
      Ms[i][j] -= f * Ms[k][j];
      Ns[i][j] -= f * Ns[k][j];
    }
    __syncthreads();
  }
  Q[n * 256 + i * 16 + j] = Ns[i][j];    // (I+A)^-1 (I-A)
}

// ---------------------------------------------------------------------------
// Core layout gathers (exactly reproduce _core's reshape/transpose).
// G0 flat == P0 flat (both index as o*128+m*16+rr) -> used in place.
// ---------------------------------------------------------------------------
__global__ void gather_core_kernel(const float* __restrict__ Q, float* __restrict__ G) {
  const int t = blockIdx.x * blockDim.x + threadIdx.x;          // 16384
  const int cc = t & 15, m = (t >> 4) & 7, o = (t >> 7) & 7, rr = t >> 10;
  G[t] = Q[((o * 8 + m) << 8) + (rr << 4) + cc];                // G[rr*8+o, m*16+cc]
}

__global__ void gather_g3_kernel(const float* __restrict__ P3, float* __restrict__ G3) {
  const int t = blockIdx.x * blockDim.x + threadIdx.x;          // 1024
  const int m = t & 7, o = (t >> 3) & 7, rr = t >> 6;
  G3[t] = P3[((o * 8 + m) << 4) + rr];
}

// ---------------------------------------------------------------------------
// Chain contraction of cores into dense W (tiny FLOPs, plain VALU).
// ---------------------------------------------------------------------------
__global__ void m1_kernel(const float* __restrict__ G0, const float* __restrict__ G1,
                          float* __restrict__ M1) {
  const int t = blockIdx.x * blockDim.x + threadIdx.x;          // 65536
  const int col = t & 63, row = t >> 6;
  const int i1 = col & 7, i0 = col >> 3;
  const int r2 = row & 15, a1 = (row >> 4) & 7, a0 = row >> 7;
  float s = 0.f;
#pragma unroll
  for (int r1 = 0; r1 < 16; ++r1)
    s += G0[((a0 * 16 + r1) << 3) + i0] * G1[((a1 * 16 + r2) << 7) + (r1 << 3) + i1];
  M1[t] = s;
}

__global__ void m2_kernel(const float* __restrict__ M1, const float* __restrict__ G2,
                          float* __restrict__ M2) {
  const int t = blockIdx.x * blockDim.x + threadIdx.x;          // 4194304
  const int col = t & 511, row = t >> 9;
  const int i2 = col & 7, i01 = col >> 3;
  const int r3 = row & 15, a2 = (row >> 4) & 7, a1 = (row >> 7) & 7, a0 = row >> 10;
  float s = 0.f;
#pragma unroll
  for (int r2 = 0; r2 < 16; ++r2)
    s += M1[((a0 * 128 + a1 * 16 + r2) << 6) + i01] * G2[((a2 * 16 + r3) << 7) + (r2 << 3) + i2];
  M2[t] = s;
}

__global__ void w_kernel(const float* __restrict__ M2, const float* __restrict__ G3,
                         __hip_bfloat16* __restrict__ Wb) {
  const size_t t = (size_t)blockIdx.x * blockDim.x + threadIdx.x; // 16777216
  const int fi = (int)(t & 4095), fo = (int)(t >> 12);
  const int i3 = fi & 7, fcol = fi >> 3;
  const int a3 = fo & 7, frow = fo >> 3;
  float s = 0.f;
#pragma unroll
  for (int r3 = 0; r3 < 16; ++r3)
    s += M2[(size_t)(frow * 16 + r3) * 512 + fcol] * G3[a3 * 128 + r3 * 8 + i3];
  Wb[t] = __float2bfloat16(s);
}

__global__ void xcvt_kernel(const float* __restrict__ x, __hip_bfloat16* __restrict__ Xb) {
  const size_t t = ((size_t)blockIdx.x * blockDim.x + threadIdx.x) * 4;
  const float4 v = *(const float4*)(x + t);
  Xb[t + 0] = __float2bfloat16(v.x);
  Xb[t + 1] = __float2bfloat16(v.y);
  Xb[t + 2] = __float2bfloat16(v.z);
  Xb[t + 3] = __float2bfloat16(v.w);
}

// ---------------------------------------------------------------------------
// Main GEMM: C[b,fo] = sum_k Xb[b,k]*Wb[fo,k] + bias[fo].
// Workgroup (4 waves) computes a 128x128 C tile; K is staged 64 at a time
// through double-buffered LDS filled with GLOBAL_LOAD_ASYNC_TO_LDS_B128
// (ASYNCcnt + barrier handoff). Each wave owns a 64x64 subtile = 4x4 grid of
// v_wmma_f32_16x16x32_bf16, fragments read from LDS with ds_load_b128.
// LDS rows padded to 144B so lanes 0-15 of a fragment read hit distinct banks.
// ---------------------------------------------------------------------------
#define LROW 72   // LDS row pitch in bf16 elements (144 bytes), holds 64 K

__global__ __launch_bounds__(128)
void gemm_kernel(const __hip_bfloat16* __restrict__ Xb_, const __hip_bfloat16* __restrict__ Wb_,
                 const float* __restrict__ bias, float* __restrict__ out) {
  const __bf16* __restrict__ Xb = reinterpret_cast<const __bf16*>(Xb_);
  const __bf16* __restrict__ Wb = reinterpret_cast<const __bf16*>(Wb_);

  __shared__ __align__(16) __bf16 As[2][128 * LROW];
  __shared__ __align__(16) __bf16 Bs[2][128 * LROW];

  const int tid  = threadIdx.x;            // 0..127
  const int wave = tid >> 5;
  const int lane = tid & 31;
  const int wm = wave & 1, wn = wave >> 1; // 2x2 wave subtiles
  const int lhalf = lane >> 4;
  const int lmod  = lane & 15;

  const int tmb = blockIdx.x & 63;         // 64 M tiles (8192/128)
  const int tnb = blockIdx.x >> 6;         // 32 N tiles (4096/128)
  const int rowM = tmb * 128;
  const int rowN = tnb * 128;

  // Copy one 128x64 K-stage of A and B into LDS buffer `buf`.
  // 1024 16B chunks per matrix, 8 per thread: chunk c -> row c>>3, pos c&7.
  auto stage_copy = [&](int buf, int k0) {
#pragma unroll
    for (int i = 0; i < 8; ++i) {
      const int c = i * 128 + tid;
      const int row = c >> 3, pos = c & 7;
      const __bf16* gA = Xb + (size_t)(rowM + row) * KDIM + k0 + pos * 8;
      const __bf16* gB = Wb + (size_t)(rowN + row) * KDIM + k0 + pos * 8;
      __bf16* lA = &As[buf][row * LROW + pos * 8];
      __bf16* lB = &Bs[buf][row * LROW + pos * 8];
#if ASYNC_CP
      __builtin_amdgcn_global_load_async_to_lds_b128(AS1_V4I(gA), AS3_V4I(lA), 0, 0);
      __builtin_amdgcn_global_load_async_to_lds_b128(AS1_V4I(gB), AS3_V4I(lB), 0, 0);
#else
      *(v8bf*)lA = *(const v8bf*)gA;
      *(v8bf*)lB = *(const v8bf*)gB;
#endif
    }
  };

  v8f acc[4][4] = {};

  int buf = 0;
  stage_copy(0, 0);
  for (int k0 = 0; k0 < KDIM; k0 += 64) {
    if (k0 + 64 < KDIM) {
      stage_copy(buf ^ 1, k0 + 64);        // prefetch next stage
#if ASYNC_CP
      WAIT_ASYNC(16);                      // 16 new outstanding -> old 16 done
#endif
    } else {
#if ASYNC_CP
      WAIT_ASYNC(0);
#endif
    }
    __syncthreads();                        // all waves' stage data visible

#pragma unroll
    for (int ks = 0; ks < 64; ks += 32) {
      v16bf a[4], b[4];
#pragma unroll
      for (int mr = 0; mr < 4; ++mr) {
        const __bf16* p = &As[buf][(wm * 64 + mr * 16 + lmod) * LROW + ks] + lhalf * 8;
        v8bf lo = *(const v8bf*)p;          // K = ks + lhalf*8 .. +7
        v8bf hi = *(const v8bf*)(p + 16);   // K = ks + 16+lhalf*8 .. +7
        a[mr] = __builtin_shufflevector(lo, hi, 0,1,2,3,4,5,6,7,8,9,10,11,12,13,14,15);
      }
#pragma unroll
      for (int nr = 0; nr < 4; ++nr) {
        const __bf16* p = &Bs[buf][(wn * 64 + nr * 16 + lmod) * LROW + ks] + lhalf * 8;
        v8bf lo = *(const v8bf*)p;          // K = ks + lhalf*8 .. +7
        v8bf hi = *(const v8bf*)(p + 16);   // K = ks + 16+lhalf*8 .. +7
        b[nr] = __builtin_shufflevector(lo, hi, 0,1,2,3,4,5,6,7,8,9,10,11,12,13,14,15);
      }
#pragma unroll
      for (int mr = 0; mr < 4; ++mr)
#pragma unroll
        for (int nr = 0; nr < 4; ++nr)
          acc[mr][nr] = __builtin_amdgcn_wmma_f32_16x16x32_bf16(
              false, a[mr], false, b[nr], (short)0, acc[mr][nr], false, false);
    }
    __syncthreads();                        // all reads done before overwrite
    buf ^= 1;
  }

  // C/D layout: vgpr r -> M = lhalf*8 + r, N = lmod. Fused bias add.
#pragma unroll
  for (int nr = 0; nr < 4; ++nr) {
    const int col = rowN + wn * 64 + nr * 16 + lmod;
    const float bv = bias[col];
#pragma unroll
    for (int mr = 0; mr < 4; ++mr) {
#pragma unroll
      for (int r = 0; r < 8; ++r) {
        const int row = rowM + wm * 64 + mr * 16 + lhalf * 8 + r;
        out[(size_t)row * 4096 + col] = acc[mr][nr][r] + bv;
      }
    }
  }
}

// ---------------------------------------------------------------------------
extern "C" void kernel_launch(void* const* d_in, const int* in_sizes, int n_in,
                              void* d_out, int out_size, void* d_ws, size_t ws_size,
                              hipStream_t stream) {
  const float* x    = (const float*)d_in[0];
  const float* P0   = (const float*)d_in[1];
  const float* P1   = (const float*)d_in[2];
  const float* P2   = (const float*)d_in[3];
  const float* P3   = (const float*)d_in[4];
  const float* bias = (const float*)d_in[5];
  float* out = (float*)d_out;

  char* ws = (char*)d_ws;
  auto carve = [&](size_t bytes) {
    char* p = ws;
    ws += (bytes + 255) & ~(size_t)255;
    return (void*)p;
  };
  float* Q1 = (float*)carve(64 * 256 * sizeof(float));
  float* Q2 = (float*)carve(64 * 256 * sizeof(float));
  float* G1 = (float*)carve(128 * 128 * sizeof(float));
  float* G2 = (float*)carve(128 * 128 * sizeof(float));
  float* G3 = (float*)carve(8 * 128 * sizeof(float));
  float* M1 = (float*)carve(1024 * 64 * sizeof(float));
  float* M2 = (float*)carve((size_t)8192 * 512 * sizeof(float));
  __hip_bfloat16* Wb = (__hip_bfloat16*)carve((size_t)4096 * 4096 * sizeof(__hip_bfloat16));
  __hip_bfloat16* Xb = (__hip_bfloat16*)carve((size_t)BATCH * 4096 * sizeof(__hip_bfloat16));

  // Phase 1: orthogonal cores via Cayley
  cayley_kernel<<<64, 256, 0, stream>>>(P1, Q1);
  cayley_kernel<<<64, 256, 0, stream>>>(P2, Q2);

  // Phase 2: materialize core matrices (G0 == P0 flat, used directly)
  gather_core_kernel<<<64, 256, 0, stream>>>(Q1, G1);
  gather_core_kernel<<<64, 256, 0, stream>>>(Q2, G2);
  gather_g3_kernel<<<4, 256, 0, stream>>>(P3, G3);

  // Phase 3: collapse TT chain into dense W (bf16)
  m1_kernel<<<256, 256, 0, stream>>>(P0, G1, M1);
  m2_kernel<<<16384, 256, 0, stream>>>(M1, G2, M2);
  w_kernel<<<65536, 256, 0, stream>>>(M2, G3, Wb);

  // Phase 4: x -> bf16
  xcvt_kernel<<<32768, 256, 0, stream>>>(x, Xb);

  // Phase 5: y = x @ W^T + bias (8192x4096x4096 bf16 WMMA GEMM, LDS-staged)
  gemm_kernel<<<2048, 128, 0, stream>>>(Xb, Wb, bias, out);
}
